// OdourPredictorEGNN_28896539967974
// MI455X (gfx1250) — matrix operand order repaired and checked by reading, not compiled, not run on hardware
//
#include <hip/hip_runtime.h>

// ---------------------------------------------------------------------------
// OdourPredictorEGNN for MI455X (gfx1250, wave32, WMMA).
// bf16 V_WMMA_F32_16X16X32_BF16 for every GEMM; dead coords branch skipped.
// Edge MLP (dominant ~104 GFLOP) fused per-node; per-block LDS weight staging
// (8x L2 traffic reduction) with global_load_async_to_lds_b128 for one slice.
// ---------------------------------------------------------------------------

typedef __attribute__((ext_vector_type(16))) __bf16        bf16x16;
typedef __attribute__((ext_vector_type(8)))  float         f32x8;
typedef __attribute__((ext_vector_type(8)))  unsigned int  u32x8;

#define BATCH   256
#define NATOM   64
#define KNN     16
#define HDIM    128
#define NODES   (BATCH * NATOM)     // 16384
#define EINP    288                 // 2H+1 = 257 padded to 9*32
#define EHID    544                 // 514 padded to 17*32
#define FUSION  288                 // 128 + 128 + 32
#define W1_STRIDE 34                // 32 + 2 pad halves -> 17-dword row stride (bank-conflict free)
#define W2_STRIDE 136               // 128 + 8 pad halves -> 16B-aligned rows for async b128

__device__ __forceinline__ unsigned short f2bf(float f) {
  unsigned int u = __float_as_uint(f);
  u += 0x7FFFu + ((u >> 16) & 1u);          // round-to-nearest-even
  return (unsigned short)(u >> 16);
}
__device__ __forceinline__ float silu_f(float x) { return x / (1.f + __expf(-x)); }

// ---------------------------------------------------------------------------
// Weight/activation fp32 -> padded bf16 conversion.
// ---------------------------------------------------------------------------
__global__ void pad_convert(const float* __restrict__ src, unsigned short* __restrict__ dst,
                            int srows, int scols, int drows, int dcols) {
  int t = blockIdx.x * blockDim.x + threadIdx.x;
  if (t >= drows * dcols) return;
  int r = t / dcols, c = t % dcols;
  float v = (r < srows && c < scols) ? src[(size_t)r * scols + c] : 0.f;
  dst[t] = f2bf(v);
}

// ---------------------------------------------------------------------------
// Generic bf16 WMMA GEMM: out = act(A[M x K] @ W[K x N] + bias) (+ residual)
// One wave computes one 16x16 output tile.
// A fragment (16x32 bf16): lane l -> row (l&15); VGPR v -> K pair
//   kb = 16*(v>>2) + 8*(l>>4) + 2*(v&3)             (ISA 7.12.2 table)
// B fragment (32x16 bf16): lane l -> col (l&15); VGPR v -> K pair kb (same as A)
// C/D (16x16 f32): VGPR r, lane l -> row r + 8*(l>>4), col l&15
// ---------------------------------------------------------------------------
__global__ void gemm_bf16_wmma(const unsigned short* __restrict__ A, int lda,
                               const unsigned short* __restrict__ W, int ldw,
                               const float* __restrict__ bias,
                               const float* __restrict__ residual, int ldr,
                               float* __restrict__ outF,
                               unsigned short* __restrict__ outB, int ldo,
                               int tilesM, int tilesN, int Nout, int kChunks, int act) {
  int wid = blockIdx.x * (blockDim.x >> 5) + (threadIdx.x >> 5);
  if (wid >= tilesM * tilesN) return;
  int lane = threadIdx.x & 31;
  int tM = wid / tilesN, tN = wid % tilesN;
  int lh = lane >> 4;                 // lane half
  int m  = tM * 16 + (lane & 15);
  int col0 = tN * 16;
  const unsigned short* arow = A + (size_t)m * lda;
  f32x8 acc;
  #pragma unroll
  for (int e = 0; e < 8; ++e) acc[e] = 0.f;

  for (int kc = 0; kc < kChunks; ++kc) {
    u32x8 ra, rb;
    #pragma unroll
    for (int v = 0; v < 8; ++v) {
      int kb = kc * 32 + ((v >> 2) << 4) + (lh << 3) + ((v & 3) << 1);
      ra[v] = *(const unsigned int*)(arow + kb);
    }
    const unsigned short* wrow = W + (size_t)(kc * 32 + lane) * ldw + col0;
    __builtin_prefetch(wrow + (size_t)32 * ldw, 0, 3);   // next K-chunk of weights
    #pragma unroll
    for (int v = 0; v < 8; ++v) {
      int kb = kc * 32 + ((v >> 2) << 4) + (lh << 3) + ((v & 3) << 1);
      const unsigned short* wp = W + (size_t)kb * ldw + col0 + (lane & 15);
      rb[v] = (unsigned int)wp[0] | ((unsigned int)wp[ldw] << 16);
    }
    acc = __builtin_amdgcn_wmma_f32_16x16x32_bf16(
        false, __builtin_bit_cast(bf16x16, ra),
        false, __builtin_bit_cast(bf16x16, rb),
        (short)0, acc, false, false);
  }

  int col = col0 + (lane & 15);
  if (col < Nout) {
    float b = bias ? bias[col] : 0.f;
    #pragma unroll
    for (int r = 0; r < 8; ++r) {
      int row = tM * 16 + r + 8 * lh;
      float x = acc[r] + b;
      if (act == 1)      x = silu_f(x);
      else if (act == 2) x = fmaxf(x, 0.f);
      if (residual) x += residual[(size_t)row * ldr + col];
      if (outF) outF[(size_t)row * ldo + col] = x;
      if (outB) outB[(size_t)row * ldo + col] = f2bf(x);
    }
  }
}

// ---------------------------------------------------------------------------
// kNN: per node, top-K smallest masked squared distances (stable ties -> low
// index, matching lax.top_k). mk = (ranking <= 10) covers mask_i & mask_j & nbhd.
// ---------------------------------------------------------------------------
__global__ void knn_kernel(const float* __restrict__ pos, const unsigned char* __restrict__ mask,
                           int* __restrict__ idx, float* __restrict__ dk,
                           float* __restrict__ mkf) {
  int node = blockIdx.x * blockDim.x + threadIdx.x;
  if (node >= NODES) return;
  int b = node >> 6, i = node & 63;
  const float* pb = pos + (size_t)b * NATOM * 3;
  float xi = pb[i * 3], yi = pb[i * 3 + 1], zi = pb[i * 3 + 2];
  bool mi = mask[b * NATOM + i] != 0;
  float rank[NATOM], dd[NATOM];
  for (int j = 0; j < NATOM; ++j) {
    float dx = xi - pb[j * 3], dy = yi - pb[j * 3 + 1], dz = zi - pb[j * 3 + 2];
    float d = dx * dx + dy * dy + dz * dz;
    dd[j] = d;
    rank[j] = (mi && mask[b * NATOM + j]) ? d : 1e5f;
  }
  for (int k = 0; k < KNN; ++k) {
    float best = 3.4e38f; int bj = 0;
    for (int j = 0; j < NATOM; ++j) {
      float r = rank[j];
      if (r < best) { best = r; bj = j; }
    }
    rank[bj] = 3.5e38f;                       // mark used
    idx[node * KNN + k] = bj;
    dk [node * KNN + k] = dd[bj];
    mkf[node * KNN + k] = (best <= 10.0f) ? 1.f : 0.f;
  }
}

// ---------------------------------------------------------------------------
// LayerNorm of h (fp32) -> bf16 into node_in[:, 0:128]. One wave per node.
// ---------------------------------------------------------------------------
__global__ void ln_kernel(const float* __restrict__ h, const float* __restrict__ g,
                          const float* __restrict__ bta, unsigned short* __restrict__ node_in) {
  int lane = threadIdx.x & 31;
  int node = blockIdx.x * (blockDim.x >> 5) + (threadIdx.x >> 5);
  const float* hp = h + (size_t)node * HDIM;
  float4 v = *(const float4*)(hp + lane * 4);
  float va[4] = {v.x, v.y, v.z, v.w};
  float s = va[0] + va[1] + va[2] + va[3];
  float sq = va[0]*va[0] + va[1]*va[1] + va[2]*va[2] + va[3]*va[3];
  #pragma unroll
  for (int off = 16; off >= 1; off >>= 1) {
    s  += __shfl_xor(s,  off, 32);
    sq += __shfl_xor(sq, off, 32);
  }
  float mu = s * (1.f / HDIM);
  float var = sq * (1.f / HDIM) - mu * mu;
  float rstd = rsqrtf(var + 1e-5f);
  unsigned short* op = node_in + (size_t)node * 256 + lane * 4;
  #pragma unroll
  for (int t = 0; t < 4; ++t) {
    int c = lane * 4 + t;
    op[t] = f2bf((va[t] - mu) * rstd * g[c] + bta[c]);
  }
}

// ---------------------------------------------------------------------------
// Fused edge MLP + masked message sum. One wave per node (16 edges = one
// 16-row WMMA tile). Per hidden slice of 32: the 8 waves of the block stage
// the e_w1 column-slice (288x32) and e_w2 row-slice (32x128) into LDS once
// (e_w2 via global_load_async_to_lds_b128), then GEMM1 -> SiLU -> LDS
// transpose bounce -> GEMM2 accumulation, all from LDS-resident weights.
// Writes masked sum m_i (bf16) into node_in[:, 128:256].
// ---------------------------------------------------------------------------
__global__ void __launch_bounds__(256, 1) edge_kernel(
    const unsigned short* __restrict__ hB, const int* __restrict__ idx,
    const float* __restrict__ dk, const float* __restrict__ mkf,
    const unsigned short* __restrict__ ew1, const float* __restrict__ eb1,
    const unsigned short* __restrict__ ew2, const float* __restrict__ eb2,
    unsigned short* __restrict__ node_in) {
  __shared__ unsigned short lds_w1[EINP * W1_STRIDE];   // 19584 B
  __shared__ unsigned short lds_w2[32 * W2_STRIDE];     //  8704 B
  __shared__ unsigned short tile[8][16 * 32];           //  8192 B hidden tiles
  int wave = threadIdx.x >> 5;
  int lane = threadIdx.x & 31;
  int node = blockIdx.x * 8 + wave;
  int b = node >> 6, i = node & 63;
  int m = lane & 15, lh = lane >> 4;
  int j = idx[node * KNN + m];
  float dm = dk[node * KNN + m];
  const unsigned short* hi = hB + (size_t)(b * NATOM + i) * HDIM;
  const unsigned short* hj = hB + (size_t)(b * NATOM + j) * HDIM;

  // Build e_in A-fragments: row m = edge m; cols [h_i | h_j | dist | 0-pad]
  bf16x16 afr[9];
  #pragma unroll
  for (int kc = 0; kc < 9; ++kc) {
    u32x8 r;
    #pragma unroll
    for (int v = 0; v < 8; ++v) {
      int kb = kc * 32 + ((v >> 2) << 4) + (lh << 3) + ((v & 3) << 1);
      unsigned int val;
      if (kb < 128)       val = *(const unsigned int*)(hi + kb);
      else if (kb < 256)  val = *(const unsigned int*)(hj + (kb - 128));
      else if (kb == 256) val = (unsigned int)f2bf(dm);   // col 256 = dist, col 257 = 0
      else                val = 0u;
      r[v] = val;
    }
    afr[kc] = __builtin_bit_cast(bf16x16, r);
  }

  float mk8[8];
  #pragma unroll
  for (int r = 0; r < 8; ++r) mk8[r] = mkf[node * KNN + r + 8 * lh];

  f32x8 acc2[8];
  #pragma unroll
  for (int t = 0; t < 8; ++t)
    #pragma unroll
    for (int e = 0; e < 8; ++e) acc2[t][e] = 0.f;

  for (int c2 = 0; c2 < EHID / 32; ++c2) {      // 17 hidden slices of 32
    __syncthreads();                            // previous slice fully consumed

    // ---- stage e_w1[:, c2*32 : c2*32+32] -> lds_w1 (288 rows x 32 halves) ----
    #pragma unroll
    for (int it = 0; it < 18; ++it) {           // 4608 dwords / 256 threads
      int id4 = threadIdx.x + 256 * it;
      int row = id4 >> 4, c4 = id4 & 15;
      *(unsigned int*)&lds_w1[row * W1_STRIDE + c4 * 2] =
          *(const unsigned int*)(ew1 + (size_t)row * EHID + c2 * 32 + c4 * 2);
    }
    // ---- stage e_w2[c2*32 : +32, :] -> lds_w2 via async 16B DMA per lane ----
    #pragma unroll
    for (int it = 0; it < 2; ++it) {            // 512 x b128 / 256 threads
      int id128 = threadIdx.x + 256 * it;
      int row = id128 >> 4, seg = id128 & 15;   // 16 x 16B per 128-half row
      unsigned long long ga =
          (unsigned long long)(const void*)(ew2 + (size_t)(c2 * 32 + row) * HDIM + seg * 8);
      unsigned int la = (unsigned int)(size_t)&lds_w2[row * W2_STRIDE + seg * 8];
      asm volatile("global_load_async_to_lds_b128 %0, %1, off"
                   :: "v"(la), "v"(ga) : "memory");
    }
    asm volatile("s_wait_asynccnt 0x0" ::: "memory");
    __syncthreads();                            // weights visible to all waves

    // ---- GEMM1: two 16-col tiles of this 32-hidden slice ----
    #pragma unroll
    for (int t = 0; t < 2; ++t) {
      int nt = c2 * 2 + t;
      f32x8 h1;
      #pragma unroll
      for (int e = 0; e < 8; ++e) h1[e] = 0.f;
      #pragma unroll
      for (int kc = 0; kc < 9; ++kc) {
        u32x8 rb;
        #pragma unroll
        for (int v = 0; v < 8; ++v) {
          int kb = kc * 32 + ((v >> 2) << 4) + (lh << 3) + ((v & 3) << 1);
          rb[v] = (unsigned int)lds_w1[kb * W1_STRIDE + t * 16 + m] |
                  ((unsigned int)lds_w1[(kb + 1) * W1_STRIDE + t * 16 + m] << 16);
        }
        h1 = __builtin_amdgcn_wmma_f32_16x16x32_bf16(
            false, afr[kc], false, __builtin_bit_cast(bf16x16, rb),
            (short)0, h1, false, false);
      }
      int colg = nt * 16 + m;
      float bias = (colg < 514) ? eb1[colg] : 0.f;
      #pragma unroll
      for (int r = 0; r < 8; ++r)
        tile[wave][(r + 8 * lh) * 32 + t * 16 + m] = f2bf(silu_f(h1[r] + bias));
    }
    asm volatile("s_wait_dscnt 0x0" ::: "memory");
    __builtin_amdgcn_wave_barrier();
    u32x8 ra;
    #pragma unroll
    for (int v = 0; v < 8; ++v) {
      int kb = ((v >> 2) << 4) + (lh << 3) + ((v & 3) << 1);
      ra[v] = *(const unsigned int*)(&tile[wave][m * 32 + kb]);
    }
    bf16x16 a2 = __builtin_bit_cast(bf16x16, ra);
    asm volatile("s_wait_dscnt 0x0" ::: "memory");   // reads done before next overwrite
    __builtin_amdgcn_wave_barrier();

    // ---- GEMM2: accumulate 16x128 messages from LDS-resident e_w2 slice ----
    #pragma unroll
    for (int t2 = 0; t2 < 8; ++t2) {
      u32x8 rb;
      #pragma unroll
      for (int v = 0; v < 8; ++v) {
        int kb = ((v >> 2) << 4) + (lh << 3) + ((v & 3) << 1);
        rb[v] = (unsigned int)lds_w2[kb * W2_STRIDE + t2 * 16 + m] |
                ((unsigned int)lds_w2[(kb + 1) * W2_STRIDE + t2 * 16 + m] << 16);
      }
      acc2[t2] = __builtin_amdgcn_wmma_f32_16x16x32_bf16(
          false, a2, false, __builtin_bit_cast(bf16x16, rb),
          (short)0, acc2[t2], false, false);
    }
  }

  // SiLU + mask + sum over 16 edge-rows -> m_i, into node_in cols 128..255
  #pragma unroll
  for (int t2 = 0; t2 < 8; ++t2) {
    float bias = eb2[t2 * 16 + m];
    float s = 0.f;
    #pragma unroll
    for (int r = 0; r < 8; ++r)
      s += silu_f(acc2[t2][r] + bias) * mk8[r];
    s += __shfl_xor(s, 16, 32);                 // combine the two row halves
    if (lh == 0)
      node_in[(size_t)node * 256 + 128 + t2 * 16 + m] = f2bf(s);
  }
}

// ---------------------------------------------------------------------------
// Masked mean pool over atoms -> g (bf16) into hg[:, 0:128].
// ---------------------------------------------------------------------------
__global__ void pool_kernel(const float* __restrict__ fea, const unsigned char* __restrict__ mask,
                            unsigned short* __restrict__ hg) {
  int b = blockIdx.x, c = threadIdx.x;          // 128 threads
  float s = 0.f; int cnt = 0;
  for (int n = 0; n < NATOM; ++n) {
    if (mask[b * NATOM + n]) { s += fea[(size_t)(b * NATOM + n) * HDIM + c]; ++cnt; }
  }
  hg[(size_t)b * FUSION + c] = f2bf(s / fmaxf((float)cnt, 1.f));
}

// ---------------------------------------------------------------------------
// Descriptor encoder: relu(descs @ de_w + de_b) -> hg[:, 256:288].
// ---------------------------------------------------------------------------
__global__ void de_kernel(const float* __restrict__ descs, const float* __restrict__ w,
                          const float* __restrict__ bias, unsigned short* __restrict__ hg) {
  int t = blockIdx.x * blockDim.x + threadIdx.x;
  if (t >= BATCH * 32) return;
  int b = t >> 5, o = t & 31;
  float s = bias[o];
  #pragma unroll
  for (int k = 0; k < 4; ++k) s += descs[b * 4 + k] * w[k * 32 + o];
  hg[(size_t)b * FUSION + 256 + o] = f2bf(fmaxf(s, 0.f));
}

// ---------------------------------------------------------------------------
extern "C" void kernel_launch(void* const* d_in, const int* in_sizes, int n_in,
                              void* d_out, int out_size, void* d_ws, size_t ws_size,
                              hipStream_t stream) {
  (void)in_sizes; (void)n_in; (void)out_size; (void)ws_size;

  const float*         x     = (const float*)d_in[0];
  const float*         pos   = (const float*)d_in[1];
  const unsigned char* mask  = (const unsigned char*)d_in[2];
  const float*         fp    = (const float*)d_in[3];
  const float*         descs = (const float*)d_in[4];
  const float* win  = (const float*)d_in[5];   const float* bin  = (const float*)d_in[6];
  const float* e_w1 = (const float*)d_in[7];   const float* e_b1 = (const float*)d_in[8];
  const float* e_w2 = (const float*)d_in[9];   const float* e_b2 = (const float*)d_in[10];
  const float* ln_g = (const float*)d_in[11];  const float* ln_b = (const float*)d_in[12];
  // d_in[13..17]: cn_scale, c_w1, c_b1, c_w2, c_b2 -> dead branch, skipped
  const float* n_w1 = (const float*)d_in[18];  const float* n_b1 = (const float*)d_in[19];
  const float* n_w2 = (const float*)d_in[20];  const float* n_b2 = (const float*)d_in[21];
  const float* fp_w = (const float*)d_in[22];  const float* fp_b = (const float*)d_in[23];
  const float* de_w = (const float*)d_in[24];  const float* de_b = (const float*)d_in[25];
  const float* h_w1 = (const float*)d_in[26];  const float* h_b1 = (const float*)d_in[27];
  const float* h_w2 = (const float*)d_in[28];  const float* h_b2 = (const float*)d_in[29];
  const float* h_w3 = (const float*)d_in[30];  const float* h_b3 = (const float*)d_in[31];

  // ---- workspace carve-up (all 256B aligned; ~48 MB total) ----
  char* ws = (char*)d_ws;
  size_t off = 0;
  auto carve = [&](size_t bytes) -> char* {
    char* p = ws + off; off += (bytes + 255) & ~(size_t)255; return p;
  };
  unsigned short* win_p  = (unsigned short*)carve((size_t)128 * 128 * 2);
  unsigned short* ew1_p  = (unsigned short*)carve((size_t)EINP * EHID * 2);
  unsigned short* ew2_p  = (unsigned short*)carve((size_t)EHID * HDIM * 2);
  unsigned short* nw1_p  = (unsigned short*)carve((size_t)256 * 256 * 2);
  unsigned short* nw2_p  = (unsigned short*)carve((size_t)256 * 128 * 2);
  unsigned short* fpw_p  = (unsigned short*)carve((size_t)2048 * 128 * 2);
  unsigned short* hw1_p  = (unsigned short*)carve((size_t)288 * 256 * 2);
  unsigned short* hw2_p  = (unsigned short*)carve((size_t)256 * 128 * 2);
  unsigned short* hw3_p  = (unsigned short*)carve((size_t)128 * 144 * 2);
  unsigned short* x_bf   = (unsigned short*)carve((size_t)NODES * 128 * 2);
  unsigned short* fp_bf  = (unsigned short*)carve((size_t)BATCH * 2048 * 2);
  float*          h_f32  = (float*)carve((size_t)NODES * HDIM * 4);
  unsigned short* h_bf   = (unsigned short*)carve((size_t)NODES * HDIM * 2);
  int*            idx    = (int*)carve((size_t)NODES * KNN * 4);
  float*          distk  = (float*)carve((size_t)NODES * KNN * 4);
  float*          mkf    = (float*)carve((size_t)NODES * KNN * 4);
  unsigned short* nodein = (unsigned short*)carve((size_t)NODES * 256 * 2);
  unsigned short* t1     = (unsigned short*)carve((size_t)NODES * 256 * 2);
  float*          fea    = (float*)carve((size_t)NODES * HDIM * 4);
  unsigned short* hg     = (unsigned short*)carve((size_t)BATCH * FUSION * 2);
  unsigned short* z1     = (unsigned short*)carve((size_t)BATCH * 256 * 2);
  unsigned short* z2     = (unsigned short*)carve((size_t)BATCH * 128 * 2);

  auto cdiv = [](int a, int b) { return (a + b - 1) / b; };
  auto conv = [&](const float* s, unsigned short* d, int sr, int sc, int dr, int dc) {
    int tot = dr * dc;
    pad_convert<<<cdiv(tot, 256), 256, 0, stream>>>(s, d, sr, sc, dr, dc);
  };

  // ---- weight / input conversion to padded bf16 ----
  conv(win,  win_p, 108, 128, 128, 128);
  conv(e_w1, ew1_p, 257, 514, EINP, EHID);
  conv(e_w2, ew2_p, 514, 128, EHID, 128);
  conv(n_w1, nw1_p, 256, 256, 256, 256);
  conv(n_w2, nw2_p, 256, 128, 256, 128);
  conv(fp_w, fpw_p, 2048, 128, 2048, 128);
  conv(h_w1, hw1_p, 288, 256, 288, 256);
  conv(h_w2, hw2_p, 256, 128, 256, 128);
  conv(h_w3, hw3_p, 128, 138, 128, 144);
  conv(x,    x_bf,  NODES, 108, NODES, 128);
  conv(fp,   fp_bf, BATCH, 2048, BATCH, 2048);

  // ---- h = x @ win + bin  (f32 + bf16 copies) ----
  gemm_bf16_wmma<<<cdiv((NODES/16)*8, 8), 256, 0, stream>>>(
      x_bf, 128, win_p, 128, bin, nullptr, 0, h_f32, h_bf, 128,
      NODES / 16, 8, 128, 4, 0);

  // ---- kNN + masks ----
  knn_kernel<<<cdiv(NODES, 256), 256, 0, stream>>>(pos, mask, idx, distk, mkf);

  // ---- LayerNorm -> node_in[:, 0:128] ----
  ln_kernel<<<NODES / 8, 256, 0, stream>>>(h_f32, ln_g, ln_b, nodein);

  // ---- fused edge MLP + masked sum -> node_in[:, 128:256] ----
  edge_kernel<<<NODES / 8, 256, 0, stream>>>(
      h_bf, idx, distk, mkf, ew1_p, e_b1, ew2_p, e_b2, nodein);

  // ---- node MLP: silu(node_in @ n_w1 + b1) @ n_w2 + b2 + h ----
  gemm_bf16_wmma<<<cdiv((NODES/16)*16, 8), 256, 0, stream>>>(
      nodein, 256, nw1_p, 256, n_b1, nullptr, 0, nullptr, t1, 256,
      NODES / 16, 16, 256, 8, 1);
  gemm_bf16_wmma<<<cdiv((NODES/16)*8, 8), 256, 0, stream>>>(
      t1, 256, nw2_p, 128, n_b2, h_f32, 128, fea, nullptr, 128,
      NODES / 16, 8, 128, 8, 0);

  // ---- masked mean pool -> hg[:, 0:128] ----
  pool_kernel<<<BATCH, 128, 0, stream>>>(fea, mask, hg);

  // ---- fingerprint encoder: relu(fp @ fp_w + b) -> hg[:, 128:256] ----
  gemm_bf16_wmma<<<cdiv((BATCH/16)*8, 8), 256, 0, stream>>>(
      fp_bf, 2048, fpw_p, 128, fp_b, nullptr, 0, nullptr, hg + 128, FUSION,
      BATCH / 16, 8, 128, 64, 2);

  // ---- descriptor encoder -> hg[:, 256:288] ----
  de_kernel<<<cdiv(BATCH * 32, 256), 256, 0, stream>>>(descs, de_w, de_b, hg);

  // ---- head MLP ----
  gemm_bf16_wmma<<<cdiv((BATCH/16)*16, 8), 256, 0, stream>>>(
      hg, FUSION, hw1_p, 256, h_b1, nullptr, 0, nullptr, z1, 256,
      BATCH / 16, 16, 256, 9, 2);
  gemm_bf16_wmma<<<cdiv((BATCH/16)*8, 8), 256, 0, stream>>>(
      z1, 256, hw2_p, 128, h_b2, nullptr, 0, nullptr, z2, 128,
      BATCH / 16, 8, 128, 8, 2);
  gemm_bf16_wmma<<<cdiv((BATCH/16)*9, 8), 256, 0, stream>>>(
      z2, 128, hw3_p, 144, h_b3, nullptr, 0, (float*)d_out, nullptr, 138,
      BATCH / 16, 9, 138, 4, 0);
}
